// _FFTConv_52776558133824
// MI455X (gfx1250) — matrix operand order repaired and checked by reading, not compiled, not run on hardware
//
#include <hip/hip_runtime.h>

typedef __attribute__((ext_vector_type(16))) _Float16 v16h;
typedef __attribute__((ext_vector_type(8)))  float    v8f;
typedef __attribute__((ext_vector_type(8)))  int      v8i;

#define NB    16
#define CIN   64
#define LSIG  16384
#define COUT  64
#define KW    128
#define TOUT  (LSIG - KW + 1)   /* 16257 valid correlation outputs */
#define NTILE 256
#define NP    192               /* pair entries per phase array: covers s[0..384] */
#define QMAX  127.0f

// ---------------- pass 0: zero the absmax slot (graph-replay safe) ----------------
__global__ void k_init(unsigned* ws) {
  if (threadIdx.x == 0 && blockIdx.x == 0) ws[0] = 0u;
}

// ---------------- pass 1: max|w| reduction over all weights ----------------
__global__ void k_absmax(const float* __restrict__ w, unsigned* __restrict__ ws, int n) {
  __shared__ float red[256];
  float m = 0.f;
  for (int idx = blockIdx.x * blockDim.x + threadIdx.x; idx < n; idx += gridDim.x * blockDim.x)
    m = fmaxf(m, fabsf(w[idx]));
  red[threadIdx.x] = m;
  __syncthreads();
  for (int s = 128; s > 0; s >>= 1) {
    if (threadIdx.x < s) red[threadIdx.x] = fmaxf(red[threadIdx.x], red[threadIdx.x + s]);
    __syncthreads();
  }
  if (threadIdx.x == 0) atomicMax(ws, __float_as_uint(red[0]));
}

// ---------------- pass 2: quantize weights to integer levels, stored f16 (exact) ----------------
__global__ void k_quant(const float* __restrict__ w, const unsigned* __restrict__ wsmax,
                        _Float16* __restrict__ q, int n) {
  float maxabs = __uint_as_float(wsmax[0]);
  float inv = (maxabs > 0.f) ? (QMAX / maxabs) : 0.f;
  int idx = blockIdx.x * blockDim.x + threadIdx.x;
  if (idx < n) q[idx] = (_Float16)rintf(w[idx] * inv);
}

// ---------------- pass 3: implicit-GEMM cross-correlation on WMMA ----------------
__launch_bounds__(256, 2)
__global__ void k_conv_wmma(const float* __restrict__ sig,
                            const _Float16* __restrict__ qw,      // [COUT][CIN][KW] f16 int levels
                            const float* __restrict__ bias,
                            const unsigned* __restrict__ wsmax,
                            float* __restrict__ out) {
  __shared__ __align__(16) unsigned sW[2][COUT * (KW / 2)];   // 2 x 16 KB weight tiles
  __shared__ unsigned sP0[2][NP];                             // (s[2j],   s[2j+1]) f16 pairs
  __shared__ unsigned sP1[2][NP];                             // (s[2j+1], s[2j+2]) f16 pairs

  const int tid  = threadIdx.x;
  const int lane = tid & 31;
  const int wave = tid >> 5;
  const int wm   = wave >> 2;      // 0..1 : 32-row Cout block
  const int wn   = wave & 3;       // 0..3 : 64-col t block
  const int half = lane >> 4;
  const int l16  = lane & 15;

  const int t0 = blockIdx.x * NTILE;
  const int b  = blockIdx.y;

  const float*    sigB = sig + (size_t)b * CIN * LSIG;
  const unsigned* qwU  = (const unsigned*)qw;

  // staging geometry (per thread)
  const int o  = tid >> 2;                       // 4 threads per output channel
  const int kd = (tid & 3) * 16;                 // dword offset within 64-dword row
  const size_t wRowOff = (size_t)o * (CIN * KW / 2) + kd;

  // ---- async weight staging: global -> LDS with no VGPR round trip (ASYNCcnt) ----
  auto issueWeightsAsync = [&](int ci, int buf) {
    const unsigned* g = qwU + wRowOff + ci * (KW / 2);
    unsigned* l = &sW[buf][o * (KW / 2) + kd];
    #pragma unroll
    for (int u = 0; u < 4; ++u) {
      unsigned long long ga = (unsigned long long)(const void*)(g + 4 * u);
      unsigned la = (unsigned)(unsigned long long)(const void*)(l + 4 * u); // LDS byte offset
      asm volatile("global_load_async_to_lds_b128 %0, %1, off"
                   :: "v"(la), "v"(ga) : "memory");
    }
  };

  // ---- signal staging: f32 loads up front (latency hidden under WMMA), pack later ----
  auto loadSignal = [&](int ci, float fr[4]) {
    const float* s = sigB + (size_t)ci * LSIG + t0;
    #pragma unroll
    for (int e = 0; e < 2; ++e) {
      int idx = tid + e * 256;                   // 384 pair-entries over 256 threads
      int phase = (idx >= NP) ? 1 : 0;
      int j = idx - phase * NP;
      int p = 2 * j + phase;
      bool v = idx < 2 * NP;
      fr[2*e+0] = (v && (t0 + p     < LSIG)) ? s[p]     : 0.f;
      fr[2*e+1] = (v && (t0 + p + 1 < LSIG)) ? s[p + 1] : 0.f;
    }
  };
  auto storeSignal = [&](int buf, const float fr[4]) {
    #pragma unroll
    for (int e = 0; e < 2; ++e) {
      int idx = tid + e * 256;
      if (idx < 2 * NP) {
        int phase = (idx >= NP) ? 1 : 0;
        int j = idx - phase * NP;
        _Float16 h0 = (_Float16)fr[2*e], h1 = (_Float16)fr[2*e+1];
        unsigned pk = (unsigned)__builtin_bit_cast(unsigned short, h0) |
                     ((unsigned)__builtin_bit_cast(unsigned short, h1) << 16);
        (phase ? sP1 : sP0)[buf][j] = pk;
      }
    }
  };

  v8f acc[2][4] = {};

  auto compute = [&](int buf) {
    // A base: weight row start is 64-dword aligned; +8*half keeps 16B alignment
    //   -> each 16x32 A fragment = two ds_load_b128 at uint4-offsets {0,1} (+ks*4, +m*256)
    const uint4* aBase =
        (const uint4*)&sW[buf][(wm * 32 + l16) * (KW / 2) + 8 * half];
    // B base: pair-array parity is a per-lane constant (parity of l16); all 8 dwords of a
    //   fragment are consecutive -> single base + compile-time immediate offsets
    const unsigned* bBase =
        ((l16 & 1) ? sP1 : sP0)[buf] + ((wn * 64 + l16) >> 1) + 8 * half;

    #pragma unroll
    for (int ks = 0; ks < 4; ++ks) {
      v16h a[2];
      #pragma unroll
      for (int m = 0; m < 2; ++m) {
        uint4 lo = aBase[m * 256 + ks * 4 + 0];   // dwords +0..3
        uint4 hi = aBase[m * 256 + ks * 4 + 1];   // dwords +4..7
        v8i ai;
        ai[0] = (int)lo.x; ai[1] = (int)lo.y; ai[2] = (int)lo.z; ai[3] = (int)lo.w;
        ai[4] = (int)hi.x; ai[5] = (int)hi.y; ai[6] = (int)hi.z; ai[7] = (int)hi.w;
        a[m] = __builtin_bit_cast(v16h, ai);
      }
      #pragma unroll
      for (int f = 0; f < 4; ++f) {               // B fragments reused across both M blocks
        v8i bi;
        #pragma unroll
        for (int v = 0; v < 8; ++v)
          bi[v] = (int)bBase[f * 8 + ks * 16 + v];  // ds_load_b32 base + imm offset
        v16h bf = __builtin_bit_cast(v16h, bi);
        #pragma unroll
        for (int m = 0; m < 2; ++m)
          acc[m][f] = __builtin_amdgcn_wmma_f32_16x16x32_f16(false, a[m], false, bf,
                                                             (short)0, acc[m][f], false, false);
      }
    }
  };

  // ---- prologue: fill buffer 0 ----
  {
    float fr[4];
    issueWeightsAsync(0, 0);
    loadSignal(0, fr);
    storeSignal(0, fr);
    asm volatile("s_wait_asynccnt 0x0" ::: "memory");
  }
  __syncthreads();

  // ---- pipelined main loop over input channels ----
  for (int ci = 0; ci < CIN; ++ci) {
    const int cur = ci & 1, nxt = cur ^ 1;
    float fr[4];
    if (ci + 1 < CIN) {
      issueWeightsAsync(ci + 1, nxt);   // async fill of other buffer
      loadSignal(ci + 1, fr);           // outstanding global loads during compute
    }
    compute(cur);
    if (ci + 1 < CIN) storeSignal(nxt, fr);
    asm volatile("s_wait_asynccnt 0x0" ::: "memory");
    __syncthreads();
  }

  // ---- epilogue: rescale (dequant) + bias + store ----
  const float scale = __uint_as_float(wsmax[0]) * (1.0f / QMAX);
  #pragma unroll
  for (int m = 0; m < 2; ++m) {
    #pragma unroll
    for (int f = 0; f < 4; ++f) {
      #pragma unroll
      for (int r = 0; r < 8; ++r) {
        int oo = wm * 32 + m * 16 + r + 8 * half;   // C/D layout: lanes16-31 carry M=r+8
        int t  = t0 + wn * 64 + f * 16 + l16;
        if (t < TOUT)
          out[((size_t)b * COUT + oo) * TOUT + t] = acc[m][f][r] * scale + bias[oo];
      }
    }
  }
}

extern "C" void kernel_launch(void* const* d_in, const int* in_sizes, int n_in,
                              void* d_out, int out_size, void* d_ws, size_t ws_size,
                              hipStream_t stream) {
  const float* sig  = (const float*)d_in[0];   // (16, 64, 16384) f32
  const float* wgt  = (const float*)d_in[1];   // (64, 64, 128) f32
  const float* bias = (const float*)d_in[2];   // (64,) f32
  float* out = (float*)d_out;                  // (16, 64, 16257) f32

  unsigned* wsmax = (unsigned*)d_ws;                 // [0]: max|w| bits
  _Float16* qw    = (_Float16*)((char*)d_ws + 16);   // 1 MiB quantized weights

  const int nW = COUT * CIN * KW;

  k_init  <<<1, 1, 0, stream>>>(wsmax);
  k_absmax<<<256, 256, 0, stream>>>(wgt, wsmax, nW);
  k_quant <<<(nW + 255) / 256, 256, 0, stream>>>(wgt, wsmax, qw, nW);

  dim3 grid((TOUT + NTILE - 1) / NTILE, NB);
  k_conv_wmma<<<grid, 256, 0, stream>>>(sig, qw, bias, wsmax, out);
}
